// RefinedCodeSwitchLoss_13494787244306
// MI455X (gfx1250) — compile-verified, matrix operand order checked by reading, not run on hardware
//
#include <hip/hip_runtime.h>
#include <math.h>

typedef __attribute__((ext_vector_type(16))) __bf16 v16bf;
typedef __attribute__((ext_vector_type(8)))  float  v8f;
typedef unsigned int u32x4 __attribute__((ext_vector_type(4)));
typedef int          i32x8 __attribute__((ext_vector_type(8)));
typedef int          i32x4 __attribute__((ext_vector_type(4)));

#define NB 2048           // batch
#define DIM 1024          // feature dim
#define SCALE 10.0f       // 1/TEMP
#define LDSL 56           // padded LDS row stride (elements) = 112 B
#define TILEB (128 * LDSL * 2)   // bytes per 128x32 padded LDS tile = 14336

// ---------- helpers ----------

static __device__ __forceinline__ unsigned int f2bf(float f) {
  unsigned int u = __float_as_uint(f);
  return (u + 0x7FFFu + ((u >> 16) & 1u)) >> 16;   // RNE to bf16 bits
}

static __device__ __forceinline__ void storeBf4(unsigned short* M, size_t row,
                                                int tid, const float* v) {
  uint2 p;
  p.x = f2bf(v[0]) | (f2bf(v[1]) << 16);
  p.y = f2bf(v[2]) | (f2bf(v[3]) << 16);
  *(uint2*)(M + row * DIM + tid * 4) = p;
}

static __device__ __forceinline__ float4 blockReduce4(float4 v, float4* red, int tid) {
  for (int m = 16; m >= 1; m >>= 1) {
    v.x += __shfl_xor(v.x, m);
    v.y += __shfl_xor(v.y, m);
    v.z += __shfl_xor(v.z, m);
    v.w += __shfl_xor(v.w, m);
  }
  __syncthreads();                       // protect red from previous use
  if ((tid & 31) == 0) red[tid >> 5] = v;
  __syncthreads();
  if (tid == 0) {
    float4 a = red[0];
    for (int i = 1; i < 8; ++i) {
      a.x += red[i].x; a.y += red[i].y; a.z += red[i].z; a.w += red[i].w;
    }
    red[0] = a;
  }
  __syncthreads();
  return red[0];
}

// TDM: DMA one 32-row x 32-col bf16 tile (row stride DIM elements in global)
// into LDS at ldsByteAddr, inserting 48B of padding after each 64B row so the
// LDS layout has a 112B row stride (matches LDSL).
static __device__ __forceinline__ void tdm_load_tile(
    const unsigned short* gbase, unsigned ldsByteAddr, unsigned rowStart, int kb) {
  unsigned long long ga = (unsigned long long)(size_t)gbase +
                          ((unsigned long long)rowStart * DIM + (unsigned)kb) * 2ull;
  // D# group 0: count=1 | lds_addr | global_addr[56:0] | type=2
  unsigned g0w1 = (unsigned)__builtin_amdgcn_readfirstlane((int)ldsByteAddr);
  unsigned g0w2 = (unsigned)__builtin_amdgcn_readfirstlane((int)(unsigned)(ga & 0xFFFFFFFFull));
  unsigned g0w3 = (unsigned)__builtin_amdgcn_readfirstlane((int)(unsigned)((ga >> 32) & 0x1FFFFFFull))
                  | 0x80000000u;  // type=2 ("image") in bits [127:126]
  u32x4 g0 = {1u, g0w1, g0w2, g0w3};
  // D# group 1:
  //  w0: workgroup_mask=0 | data_size=1 (2B) | pad_enable | pad_interval=3 (64B)
  //      | pad_amount=11 (48B)
  //  w1: atomic_barrier_addr=0 | tensor_dim0[15:0]=1024
  //  w2: tensor_dim0[31:16]=0  | tensor_dim1[15:0]=8192
  //  w3: tensor_dim1[31:16]=0  | tile_dim0=32
  //  w4: tile_dim1=32 | tile_dim2=0
  //  w5: tensor_dim0_stride[31:0]=1024 ; w6,w7: 0
  i32x8 g1 = { (int)((1u << 16) | (1u << 20) | (3u << 22) | (11u << 25)),
               (int)(1024u << 16),
               (int)(8192u << 16),
               (int)(32u << 16),
               32,
               1024,
               0, 0 };
  i32x4 z4 = {0, 0, 0, 0};
#if defined(__clang_major__) && (__clang_major__ >= 23)
  i32x8 z8 = {0, 0, 0, 0, 0, 0, 0, 0};
  __builtin_amdgcn_tensor_load_to_lds(g0, g1, z4, z4, z8, 0);
#else
  __builtin_amdgcn_tensor_load_to_lds(g0, g1, z4, z4, 0);
#endif
}

// ---------- kernels ----------

__global__ void zero_kernel(float* p, int n) {
  int i = blockIdx.x * 256 + threadIdx.x;
  if (i < n) p[i] = 0.f;
}

// One block per batch row: norms, fp32 diagonal dots (numerators), cs-reg
// norms, and bf16 emission of the stacked matrix M = [e; k; et; kt].
__global__ __launch_bounds__(256) void norm_diag_reg_kernel(
    const float* __restrict__ eg, const float* __restrict__ etg,
    const float* __restrict__ ktg, const float* __restrict__ kg,
    const float* __restrict__ ratio, unsigned short* __restrict__ M,
    float* __restrict__ rowNum, float* __restrict__ regSum) {
  __shared__ float4 red[8];
  const int b = blockIdx.x;
  const int tid = threadIdx.x;
  const size_t base = (size_t)b * 256 + tid;

  float4 e4  = ((const float4*)eg)[base];
  float4 k4  = ((const float4*)kg)[base];
  float4 et4 = ((const float4*)etg)[base];
  float4 kt4 = ((const float4*)ktg)[base];
  float e[4]  = {e4.x,  e4.y,  e4.z,  e4.w};
  float k[4]  = {k4.x,  k4.y,  k4.z,  k4.w};
  float et[4] = {et4.x, et4.y, et4.z, et4.w};
  float kt[4] = {kt4.x, kt4.y, kt4.z, kt4.w};

  float4 p = {0.f, 0.f, 0.f, 0.f};
  for (int i = 0; i < 4; ++i) {
    p.x += e[i] * e[i];  p.y += k[i] * k[i];
    p.z += et[i] * et[i]; p.w += kt[i] * kt[i];
  }
  float4 s = blockReduce4(p, red, tid);
  float ie = rsqrtf(s.x), ik = rsqrtf(s.y), iet = rsqrtf(s.z), ikt = rsqrtf(s.w);
  for (int i = 0; i < 4; ++i) { e[i] *= ie; k[i] *= ik; et[i] *= iet; kt[i] *= ikt; }

  p = make_float4(0.f, 0.f, 0.f, 0.f);
  for (int i = 0; i < 4; ++i) {
    p.x += e[i] * k[i];  p.y += e[i] * et[i];
    p.z += e[i] * kt[i]; p.w += k[i] * et[i];
  }
  float4 d1 = blockReduce4(p, red, tid);

  const float rr = ratio[b];
  p = make_float4(0.f, 0.f, 0.f, 0.f);
  for (int i = 0; i < 4; ++i) {
    p.x += k[i] * kt[i];
    float de = et[i] - (rr * e[i] + (1.f - rr) * k[i]);
    float dk = kt[i] - ((1.f - rr) * e[i] + rr * k[i]);
    p.y += de * de;
    p.z += dk * dk;
  }
  float4 d2 = blockReduce4(p, red, tid);

  if (tid == 0) {
    float nume = __expf(d1.x * SCALE) + __expf(d1.y * SCALE) + __expf(d1.z * SCALE);
    float numk = __expf(d1.x * SCALE) + __expf(d1.w * SCALE) + __expf(d2.x * SCALE);
    rowNum[b] = nume;
    rowNum[NB + b] = numk;
    atomicAdd(regSum, sqrtf(d2.y) + sqrtf(d2.z));
  }

  storeBf4(M, (size_t)b,          tid, e);
  storeBf4(M, (size_t)b + 2048,   tid, k);
  storeBf4(M, (size_t)b + 4096,   tid, et);
  storeBf4(M, (size_t)b + 6144,   tid, kt);
}

// Tiled bf16 WMMA GEMM  S = E(4096x1024) * M^T(1024x8192), fused with
// exp(S/T), diagonal exclusion and per-row "negatives" reduction.
// Block tile 128x128, 8 waves, each wave 32 rows x 64 cols (2x4 WMMA tiles).
// Tile staging: per-wave TDM tensor_load_to_lds, double buffered.
__global__ __launch_bounds__(256) void gemm_exp_rowsum_kernel(
    const unsigned short* __restrict__ M, float* __restrict__ rowNeg) {
  __shared__ __align__(16) unsigned short As[2 * 128 * LDSL];
  __shared__ __align__(16) unsigned short Bs[2 * 128 * LDSL];

  const int tid = threadIdx.x;
  const int colTile = blockIdx.x;     // 0..63
  const int rowTile = blockIdx.y;     // 0..31
  const int wave = tid >> 5;
  const int lane = tid & 31;
  const int half = lane >> 4;         // 0 or 1
  const int l15 = lane & 15;
  const int waveRow = wave & 3;       // 4 groups of 32 rows
  const int waveCol = wave >> 2;      // 2 groups of 64 cols
  const int rowBase = rowTile * 128;
  const int colBase = colTile * 128;
  const int koffA = half ? 8 : 0;     // ISA 16-bit A/B-fragment K placement

  // Per-wave TDM assignment: waves 0-3 load A slices, waves 4-7 load B slices.
  const int isB = wave >> 2;
  const int slice = wave & 3;
  const unsigned rowStart = (unsigned)((isB ? colBase : rowBase) + slice * 32);
  const unsigned ldsSlice =
      (unsigned)(size_t)(isB ? (unsigned short*)Bs : (unsigned short*)As) +
      (unsigned)(slice * 32 * LDSL * 2);

  v8f acc[2][4] = {};

  const int nIter = DIM / 32;   // 32 K-steps

  // prologue: fill buffer 0
  tdm_load_tile(M, ldsSlice, rowStart, 0);

  for (int it = 0; it < nIter; ++it) {
    const int cur = it & 1;
    if (it + 1 < nIter) {
      // DMA next tile into the other buffer (its readers finished last iter)
      tdm_load_tile(M, ldsSlice + (unsigned)((1 - cur) * TILEB),
                    rowStart, (it + 1) * 32);
      __builtin_amdgcn_s_wait_tensorcnt(1);   // current tile's DMA complete
    } else {
      __builtin_amdgcn_s_wait_tensorcnt(0);
    }
    __syncthreads();   // all waves' DMAs for buffer `cur` are done

    union Frag { uint4 q[2]; v16bf v; } fa[2], fb[4];
    const int abase = cur * 128 * LDSL;
#pragma unroll
    for (int m = 0; m < 2; ++m) {
      int rr = abase + (waveRow * 32 + m * 16 + l15) * LDSL;
      fa[m].q[0] = *(const uint4*)(&As[rr + koffA]);        // K 0-7 / 8-15
      fa[m].q[1] = *(const uint4*)(&As[rr + koffA + 16]);   // K 16-23 / 24-31
    }
#pragma unroll
    for (int n = 0; n < 4; ++n) {
      int cc = abase + (waveCol * 64 + n * 16 + l15) * LDSL;
      fb[n].q[0] = *(const uint4*)(&Bs[cc + koffA]);
      fb[n].q[1] = *(const uint4*)(&Bs[cc + koffA + 16]);
    }
#pragma unroll
    for (int m = 0; m < 2; ++m)
#pragma unroll
      for (int n = 0; n < 4; ++n)
        acc[m][n] = __builtin_amdgcn_wmma_f32_16x16x32_bf16(
            false, fa[m].v, false, fb[n].v, (short)0, acc[m][n], false, false);

    __syncthreads();   // everyone done reading buffer `cur`; safe to overwrite
  }

  // epilogue: exp(S/T), drop diagonal entries, reduce cols -> per-row neg sums
#pragma unroll
  for (int m = 0; m < 2; ++m) {
    const int rb = rowBase + waveRow * 32 + m * 16 + half * 8;
    float rs[8];
#pragma unroll
    for (int j = 0; j < 8; ++j) {
      const int gRow = rb + j;
      const int bq = gRow & (NB - 1);
      float s = 0.f;
#pragma unroll
      for (int n = 0; n < 4; ++n) {
        int gCol = colBase + waveCol * 64 + n * 16 + l15;
        float ev = __expf(acc[m][n][j] * SCALE);
        if ((gCol & (NB - 1)) == bq) ev = 0.f;   // exclude diagonal
        s += ev;
      }
      s += __shfl_xor(s, 1);
      s += __shfl_xor(s, 2);
      s += __shfl_xor(s, 4);
      s += __shfl_xor(s, 8);    // reduce across the 16 lanes sharing this row
      rs[j] = s;
    }
    if (l15 == 0) {             // lanes 0 and 16 own 8 rows each
#pragma unroll
      for (int j = 0; j < 8; ++j) atomicAdd(&rowNeg[rb + j], rs[j]);
    }
  }
}

__global__ __launch_bounds__(256) void finalize_kernel(
    const float* __restrict__ rowNum, const float* __restrict__ rowNeg,
    const float* __restrict__ regSum, float* __restrict__ out) {
  __shared__ float red[8];
  const int tid = threadIdx.x;
  float s = 0.f;
  for (int b = tid; b < NB; b += 256) {
    float ne = rowNum[b],      ge = rowNeg[b];
    float nk = rowNum[NB + b], gk = rowNeg[NB + b];
    s += -__logf(ne / (ne + ge));
    s += -__logf(nk / (nk + gk));
  }
  for (int m = 16; m >= 1; m >>= 1) s += __shfl_xor(s, m);
  if ((tid & 31) == 0) red[tid >> 5] = s;
  __syncthreads();
  if (tid == 0) {
    float t = 0.f;
    for (int i = 0; i < 8; ++i) t += red[i];
    float contrastive = t / (2.0f * NB);
    float cs = regSum[0] / (float)NB;
    out[0] = contrastive + 0.5f * cs;
    out[1] = contrastive;
    out[2] = cs;
  }
}

// ---------- launcher ----------

extern "C" void kernel_launch(void* const* d_in, const int* in_sizes, int n_in,
                              void* d_out, int out_size, void* d_ws, size_t ws_size,
                              hipStream_t stream) {
  (void)in_sizes; (void)n_in; (void)out_size; (void)ws_size;
  const float* e  = (const float*)d_in[0];   // english
  const float* et = (const float*)d_in[1];   // etok
  const float* kt = (const float*)d_in[2];   // ktoe
  const float* k  = (const float*)d_in[3];   // korean
  const float* r  = (const float*)d_in[4];   // cs_ratios
  float* out = (float*)d_out;

  char* ws = (char*)d_ws;
  unsigned short* M = (unsigned short*)ws;                   // 8192x1024 bf16 (16 MB)
  float* rowNeg = (float*)(ws + (size_t)8192 * 1024 * 2);    // 4096 f32
  float* rowNum = rowNeg + 4096;                             // 4096 f32
  float* regSum = rowNum + 4096;                             // 1 f32

  // clear rowNeg + rowNum + regSum in one contiguous span (8193 floats)
  zero_kernel<<<(8193 + 255) / 256, 256, 0, stream>>>(rowNeg, 8193);

  norm_diag_reg_kernel<<<NB, 256, 0, stream>>>(e, et, kt, k, r, M, rowNum, regSum);

  dim3 grid(64, 32);
  gemm_exp_rowsum_kernel<<<grid, 256, 0, stream>>>(M, rowNeg);

  finalize_kernel<<<1, 256, 0, stream>>>(rowNum, rowNeg, regSum, out);
}
